// CharRNN_86792699118067
// MI455X (gfx1250) — matrix operand order repaired and checked by reading, not compiled, not run on hardware
//
#include <hip/hip_runtime.h>
#include <hip/hip_bf16.h>
#include <math.h>

// CDNA5 / gfx1250: wave32, WMMA 16x16x32 bf16 (f32 accum).
typedef __bf16 bf16_t;
typedef bf16_t v8bf  __attribute__((ext_vector_type(8)));
typedef bf16_t v16bf __attribute__((ext_vector_type(16)));
typedef float  v8f   __attribute__((ext_vector_type(8)));

#define B_  256   // batch
#define L_  1024  // sequence length
#define V_  96    // vocab
#define E_  64    // embed
#define H_  256   // hidden

// ---------------------------------------------------------------------------
// Hardware tanh (gfx1250 V_TANH_F32) — libm tanhf is ~30 VALU + a divergent
// branch, directly on the scan's serial critical path.
// ---------------------------------------------------------------------------
__device__ __forceinline__ float fast_tanh(float x) {
#if __has_builtin(__builtin_amdgcn_tanhf)
    return __builtin_amdgcn_tanhf(x);
#else
    float r;
    asm volatile("v_tanh_f32 %0, %1\n\tv_nop" : "=v"(r) : "v"(x));
    return r;
#endif
}

// ---------------------------------------------------------------------------
// Prep kernels: build the small LDS-resident operands in global scratch.
// ---------------------------------------------------------------------------

// P[v][n] = (emb @ Wx)[v][n] + b_ih[n]   (96 x 256, f32)
// Collapses the entire "ble,eh" einsum + embedding gather into a 96-row table.
__global__ void prep_P(const float* __restrict__ emb, const float* __restrict__ W_ih,
                       const float* __restrict__ b_ih, float* __restrict__ Pg) {
    int v = blockIdx.x;      // 0..95
    int n = threadIdx.x;     // 0..255
    float acc = b_ih[n];
#pragma unroll 8
    for (int e = 0; e < E_; ++e)
        acc += emb[v * E_ + e] * W_ih[e * H_ + n];
    Pg[v * H_ + n] = acc;
}

// WhT[n][k] = bf16( W_ih[64+k][n] )  (256 x 256 bf16, K contiguous per row)
__global__ void prep_whT(const float* __restrict__ W_ih, bf16_t* __restrict__ whT) {
    int n = blockIdx.x;      // 0..255
    int k = threadIdx.x;     // 0..255
    whT[n * H_ + k] = (bf16_t)W_ih[(E_ + k) * H_ + n];
}

// WhoT[n][k] = bf16( W_ho[k][n] )  (96 x 256 bf16)
__global__ void prep_whoT(const float* __restrict__ W_ho, bf16_t* __restrict__ whoT) {
    int n = blockIdx.x;      // 0..95
    int k = threadIdx.x;     // 0..255
    whoT[n * H_ + k] = (bf16_t)W_ho[k * V_ + n];
}

// xT[t][b] = x[b][t]  — time-major tokens so each step's 16 tokens are
// contiguous; a lane grabs its 8 with two global_load_b128.
__global__ void prep_xT(const int* __restrict__ x, int* __restrict__ xT) {
    int t = blockIdx.x;      // 0..1023
    int b = threadIdx.x;     // 0..255
    xT[t * B_ + b] = x[b * L_ + t];
}

// ---------------------------------------------------------------------------
// Fragment loaders (LDS -> VGPR), per the CDNA5 ISA VGPR layouts.
// ---------------------------------------------------------------------------

// 16-bit A 16x32: lane = (m = lane&15, khalf = lane>>4);
// VGPRs 0-3 hold K = khalf*8 + 0..7, VGPRs 4-7 hold K = khalf*8 + 16..23.
__device__ __forceinline__ v16bf load_fragA(const bf16_t* rowPtr, int k0, int khalf) {
    union { v16bf v; v8bf h[2]; } u;
    u.h[0] = *(const v8bf*)(rowPtr + k0 + khalf * 8);
    u.h[1] = *(const v8bf*)(rowPtr + k0 + khalf * 8 + 16);
    return u.v;
}

// 16-bit B 32x16: lane = (n = lane&15, khalf = lane>>4);
// VGPRs 0-3 hold K = khalf*8 + 0..7, VGPRs 4-7 hold K = khalf*8 + 16..23.
__device__ __forceinline__ v16bf load_fragB(const bf16_t* colPtr, int k0, int khalf) {
    union { v16bf v; v8bf h[2]; } u;
    u.h[0] = *(const v8bf*)(colPtr + k0 + khalf * 8);
    u.h[1] = *(const v8bf*)(colPtr + k0 + khalf * 8 + 16);
    return u.v;
}

// ---------------------------------------------------------------------------
// Main fused scan kernel. grid = 16 workgroups (16 batch rows each, one WGP
// apiece), block = 512 threads = 16 wave32s. Wave w owns output columns
// n0 = 16*w of the 16x256 H tile; waves 0..5 also own one 16x16 logits tile.
// Hidden state double-buffered in LDS -> one barrier per timestep; the xproj
// gather is software-pipelined one step ahead of its use.
// ---------------------------------------------------------------------------
__launch_bounds__(512, 1)
__global__ void rnn_scan(const int* __restrict__ xT,         // (L,B) tokens
                         const float* __restrict__ hidden0,  // (B,H) initial h
                         const float* __restrict__ Pg,       // (V,H) f32
                         const bf16_t* __restrict__ whT_g,   // (H,H) bf16 [n][k]
                         const bf16_t* __restrict__ whoT_g,  // (V,H) bf16 [n][k]
                         const float* __restrict__ b_ho,     // (V,)
                         float* __restrict__ logits,         // (B,L,V)
                         float* __restrict__ hidden_out) {   // (B,H)
    __shared__ __align__(16) bf16_t sWhT[H_ * H_];  // 128 KB
    __shared__ __align__(16) bf16_t sWho[V_ * H_];  //  48 KB
    __shared__ __align__(16) float  sP[V_ * H_];    //  96 KB
    __shared__ __align__(16) bf16_t sHa[16 * H_];   //   8 KB  h state (ping)
    __shared__ __align__(16) bf16_t sHb[16 * H_];   //   8 KB  h state (pong)

    const int tid    = threadIdx.x;
    const int wave   = tid >> 5;     // 0..15
    const int lane   = tid & 31;
    const int batch0 = blockIdx.x * 16;

    // --- cooperative LDS fill (16B chunks) ---
    {
        const uint4* src = (const uint4*)whT_g;
        uint4*       dst = (uint4*)sWhT;
        for (int i = tid; i < H_ * H_ * 2 / 16; i += 512) dst[i] = src[i];
        src = (const uint4*)whoT_g; dst = (uint4*)sWho;
        for (int i = tid; i < V_ * H_ * 2 / 16; i += 512) dst[i] = src[i];
        src = (const uint4*)Pg;     dst = (uint4*)sP;
        for (int i = tid; i < V_ * H_ * 4 / 16; i += 512) dst[i] = src[i];
        // initial hidden state (reference passes zeros, but honor the input)
        for (int i = tid; i < 16 * H_; i += 512)
            sHa[i] = (bf16_t)hidden0[batch0 * H_ + i];
    }
    __syncthreads();

    const int m     = lane & 15;   // A-fragment row
    const int khalf = lane >> 4;
    const int n     = lane & 15;   // C-fragment column
    const int mhalf = lane >> 4;   // C-fragment row-half
    const int n0    = wave * 16;   // this wave's H output-column base

    const bf16_t* bCol = sWhT + (n0 + n) * H_;

    const bool doLog = (wave < 6);                 // 6 tiles cover V=96
    const bf16_t* boCol = sWho + (n0 + n) * H_;    // valid iff doLog
    const float bho = doLog ? b_ho[n0 + n] : 0.0f;

    // 8 tokens this lane needs per step are contiguous in xT (16B aligned).
    const int* tokPtr = xT + batch0 + 8 * mhalf;

    bf16_t* cur = sHa;   // holds h_{t-1}
    bf16_t* nxt = sHb;   // receives h_t

    // xproj gather for a given step: 2x global_load_b128 + 8x ds_load_b32.
    auto gather = [&](int t, float pv[8]) {
        int4 a4 = *(const int4*)(tokPtr + t * B_);
        int4 b4 = *(const int4*)(tokPtr + t * B_ + 4);
        int tk[8] = {a4.x, a4.y, a4.z, a4.w, b4.x, b4.y, b4.z, b4.w};
#pragma unroll
        for (int r = 0; r < 8; ++r)
            pv[r] = sP[tk[r] * H_ + n0 + n];
    };

    float pv[8];
    gather(0, pv);   // prologue: step-0 gather

    for (int t = 0; t < L_; ++t) {
        // ---- h_pre tile: (16x256 bf16) @ (256x16 bf16), two indep chains ----
        const bf16_t* aRow = cur + m * H_;
        v8f acc0 = {0.f, 0.f, 0.f, 0.f, 0.f, 0.f, 0.f, 0.f};
        v8f acc1 = {0.f, 0.f, 0.f, 0.f, 0.f, 0.f, 0.f, 0.f};
#pragma unroll
        for (int k0 = 0; k0 < H_; k0 += 64) {
            acc0 = __builtin_amdgcn_wmma_f32_16x16x32_bf16(
                false, load_fragA(aRow, k0, khalf),
                false, load_fragB(bCol, k0, khalf), (short)0, acc0, false, false);
            acc1 = __builtin_amdgcn_wmma_f32_16x16x32_bf16(
                false, load_fragA(aRow, k0 + 32, khalf),
                false, load_fragB(bCol, k0 + 32, khalf), (short)0, acc1, false, false);
        }

        // ---- h_t = tanh(h_pre + xproj), hardware V_TANH_F32 ----
        float hval[8];
#pragma unroll
        for (int r = 0; r < 8; ++r)
            hval[r] = fast_tanh(acc0[r] + acc1[r] + pv[r]);

        // ---- publish h_t into the other buffer ----
#pragma unroll
        for (int r = 0; r < 8; ++r) {
            int row = r + 8 * mhalf;
            nxt[row * H_ + n0 + n] = (bf16_t)hval[r];
        }
        if (t == L_ - 1) {
#pragma unroll
            for (int r = 0; r < 8; ++r) {
                int row = r + 8 * mhalf;
                hidden_out[(batch0 + row) * H_ + n0 + n] = hval[r];
            }
        }

        // ---- prefetch next step's xproj; hides behind barrier + logits ----
        float pvn[8];
        gather((t + 1 < L_) ? t + 1 : t, pvn);

        __syncthreads();   // h_t visible; all reads of `cur` complete

        // ---- fused logits: (16x256) @ (256x96) from the fresh state ----
        if (doLog) {
            const bf16_t* aRowN = nxt + m * H_;
            v8f acc2 = {0.f, 0.f, 0.f, 0.f, 0.f, 0.f, 0.f, 0.f};
            v8f acc3 = {0.f, 0.f, 0.f, 0.f, 0.f, 0.f, 0.f, 0.f};
#pragma unroll
            for (int k0 = 0; k0 < H_; k0 += 64) {
                acc2 = __builtin_amdgcn_wmma_f32_16x16x32_bf16(
                    false, load_fragA(aRowN, k0, khalf),
                    false, load_fragB(boCol, k0, khalf), (short)0, acc2, false, false);
                acc3 = __builtin_amdgcn_wmma_f32_16x16x32_bf16(
                    false, load_fragA(aRowN, k0 + 32, khalf),
                    false, load_fragB(boCol, k0 + 32, khalf), (short)0, acc3, false, false);
            }
#pragma unroll
            for (int r = 0; r < 8; ++r) {
                int row = r + 8 * mhalf;
                __builtin_nontemporal_store(
                    acc2[r] + acc3[r] + bho,
                    &logits[((size_t)(batch0 + row) * L_ + t) * V_ + n0 + n]);
            }
        }

#pragma unroll
        for (int r = 0; r < 8; ++r) pv[r] = pvn[r];

        // swap ping-pong buffers; next step writes the buffer last read above
        bf16_t* tmp = cur; cur = nxt; nxt = tmp;
    }
}

// ---------------------------------------------------------------------------
extern "C" void kernel_launch(void* const* d_in, const int* in_sizes, int n_in,
                              void* d_out, int out_size, void* d_ws, size_t ws_size,
                              hipStream_t stream) {
    const int*   x      = (const int*)d_in[0];    // (256,1024) tokens
    const float* hidden = (const float*)d_in[1];  // (256,256)
    const float* emb    = (const float*)d_in[2];  // (96,64)
    const float* W_ih   = (const float*)d_in[3];  // (320,256)
    const float* b_ih   = (const float*)d_in[4];  // (256,)
    const float* W_ho   = (const float*)d_in[5];  // (256,96)
    const float* b_ho   = (const float*)d_in[6];  // (96,)

    float* logits     = (float*)d_out;                      // 256*1024*96
    float* hidden_out = logits + (size_t)B_ * L_ * V_;      // 256*256

    // scratch layout (all 16B aligned)
    char*   ws   = (char*)d_ws;
    float*  Pg   = (float*)ws;                              //  98304 B
    bf16_t* whT  = (bf16_t*)(ws + 98304);                   // 131072 B
    bf16_t* whoT = (bf16_t*)(ws + 98304 + 131072);          //  49152 B
    int*    xTg  = (int*)(ws + 98304 + 131072 + 49152);     // 1048576 B

    prep_P  <<<V_,  H_, 0, stream>>>(emb, W_ih, b_ih, Pg);
    prep_whT<<<H_,  H_, 0, stream>>>(W_ih, whT);
    prep_whoT<<<V_, H_, 0, stream>>>(W_ho, whoT);
    prep_xT <<<L_,  B_, 0, stream>>>(x, xTg);

    rnn_scan<<<B_ / 16, 512, 0, stream>>>(xTg, hidden, Pg, whT, whoT, b_ho,
                                          logits, hidden_out);
}
